// AttPCB_29583734734873
// MI455X (gfx1250) — matrix operand order repaired and checked by reading, not compiled, not run on hardware
//
#include <hip/hip_runtime.h>
#include <hip/hip_bf16.h>

// ---------------------------------------------------------------------------
// AttPCB forward for MI455X (gfx1250), bf16 WMMA + async global->LDS DMA.
//   C=2048 D=512 P=6 G=8 FD=64 N=4096
// Pipeline:
//   K0a: convert wq/wk/wv/wo fp32 -> bf16 into workspace
//   K0b: transpose+convert x [N,C,P] fp32 -> xbf [N,P,C] bf16 (LDS-staged)
//   K1 : per (16-batch, group) block: QKV GEMM (v_wmma_f32_16x16x32_bf16,
//        double-buffered LDS, A+B tiles via global_load_async_to_lds_b128),
//        bias, 6x6 attention softmax in LDS, attn@V -> bf16 AO[N,P,D]
//   K2 : output projection GEMM (wmma, A+B async) + bias + residual -> fp32
// Workspace (ushort*):
//   [0, 4*DC)                 wq,wk,wv,wo bf16
//   [4*DC, +N*P*D)            AO  [N,P,D] bf16
//   [4*DC+N*P*D, +N*P*C)      xbf [N,P,C] bf16
// ---------------------------------------------------------------------------

#define C_DIM 2048
#define D_DIM 512
#define P_DIM 6
#define G_DIM 8
#define FD_DIM 64
#define N_DIM 4096
#define NB 16                 // batch elements per block
#define COLS (NB * P_DIM)     // 96 token-columns per block
#define DC (D_DIM * C_DIM)    // elements per weight matrix

typedef __attribute__((ext_vector_type(8)))  float  v8f;
typedef __attribute__((ext_vector_type(16))) __bf16 v16bf;

union Frag { uint4 u[2]; v16bf v; };

__device__ __forceinline__ unsigned short f2bf(float f) {
  unsigned int u = __float_as_uint(f);
  u += 0x7FFFu + ((u >> 16) & 1u);      // round-to-nearest-even
  return (unsigned short)(u >> 16);
}

// Async 16B global -> LDS copy (ASYNCcnt-tracked, bypasses VGPRs).
// Generic pointer low 32 bits == LDS offset (aperture truncation rule).
__device__ __forceinline__ void async_copy_b128(const void* gsrc, void* ldst) {
  unsigned lds = (unsigned)(size_t)ldst;
  unsigned long long ga = (unsigned long long)(size_t)gsrc;
  asm volatile("global_load_async_to_lds_b128 %0, %1, off"
               :: "v"(lds), "v"(ga) : "memory");
}
__device__ __forceinline__ void wait_async0() {
  asm volatile("s_wait_asynccnt 0x0" ::: "memory");
}

// --------------------------- K0a: weight conversion ------------------------
__global__ void cvt_weights(const float* __restrict__ wq, const float* __restrict__ wk,
                            const float* __restrict__ wv, const float* __restrict__ wo,
                            unsigned short* __restrict__ wbf) {
  const int total = 4 * DC;
  for (int idx = blockIdx.x * blockDim.x + threadIdx.x; idx < total;
       idx += gridDim.x * blockDim.x) {
    int m = idx / DC;
    int r = idx - m * DC;
    const float* s = (m == 0) ? wq : ((m == 1) ? wk : ((m == 2) ? wv : wo));
    wbf[idx] = f2bf(s[r]);
  }
}

// -------------- K0b: x [N,C,P] fp32 -> xbf [N,P,C] bf16 (transpose) --------
// one block per batch element; LDS-staged so reads & writes are coalesced
__global__ __launch_bounds__(256) void cvt_x(const float* __restrict__ x,
                                             unsigned short* __restrict__ xbf) {
  __shared__ __align__(16) unsigned short t[P_DIM][C_DIM];   // 24 KB
  const int tid = threadIdx.x;
  const int n   = blockIdx.x;
  const float* xs = x + (size_t)n * C_DIM * P_DIM;
  for (int idx = tid; idx < C_DIM * P_DIM; idx += 256) {     // coalesced fp32 reads
    int c = idx / P_DIM, p = idx - c * P_DIM;
    t[p][c] = f2bf(xs[idx]);
  }
  __syncthreads();
  unsigned short* xd = xbf + (size_t)n * P_DIM * C_DIM;
  for (int idx = tid; idx < (C_DIM * P_DIM) / 8; idx += 256) {  // b128 writes
    int p = idx / (C_DIM / 8), c8 = idx - p * (C_DIM / 8);
    *reinterpret_cast<uint4*>(xd + (size_t)p * C_DIM + c8 * 8) =
        *reinterpret_cast<const uint4*>(&t[p][c8 * 8]);
  }
}

// ------------------- K1: QKV projection + attention (fused) ----------------
// grid = (N/16, G), block = 384 (12 waves; wave w -> matrix w>>2, M-tile w&3)
__global__ __launch_bounds__(384) void attpcb_qkv_attn(
    const unsigned short* __restrict__ xbf, // bf16 [N,P,C]
    const unsigned short* __restrict__ wbf, // wq|wk|wv bf16, each [D,C]
    const float* __restrict__ bq, const float* __restrict__ bk,
    const float* __restrict__ bv,
    unsigned short* __restrict__ ao)        // out: bf16 [N,P,D]
{
  __shared__ __align__(16) unsigned short Abuf[2][12][16][32]; // dbl-buffered A
  __shared__ __align__(16) unsigned short Bbuf[2][COLS][32];   // dbl-buffered X^T
  __shared__ float qkv[3][FD_DIM][COLS];                       // q,k,v fp32
  __shared__ __align__(16) unsigned short aot[COLS][72];       // AO staging (padded)
  __shared__ float biasl[3][FD_DIM];
  __shared__ float sc[NB][P_DIM][P_DIM];                       // scores / attn

  const int tid  = threadIdx.x;
  const int wave = tid >> 5;
  const int lane = tid & 31;
  const int n0   = blockIdx.x * NB;
  const int g    = blockIdx.y;

  if (tid < 3 * FD_DIM) {
    int m = tid >> 6, f = tid & 63;
    const float* b = (m == 0) ? bq : ((m == 1) ? bk : bv);
    biasl[m][f] = b[g * FD_DIM + f];
  }

  const int mW   = wave >> 2;           // 0=q 1=k 2=v
  const int mt   = wave & 3;            // M-tile within the 64 group rows
  const int arow = lane & 15;
  const int kbA  = (lane >> 4) * 8;     // A frag: K {kbA..+7, kbA+16..+23}
  const int kbB  = (lane >> 4) * 8;     // B frag: K {kbB..+7, kbB+16..+23}

  auto load_tiles = [&](int step, int buf) {
    const int k0 = step * 32;
    // A: 12 tiles x 16 rows x 4 segs of 8 bf16 -> async b128 (2 per thread)
    for (int idx = tid; idx < 12 * 16 * 4; idx += 384) {
      int tile = idx >> 6, rem = idx & 63, row = rem >> 2, seg = rem & 3;
      int m = tile >> 2;
      int d = g * FD_DIM + (tile & 3) * 16 + row;
      async_copy_b128(wbf + (size_t)m * DC + (size_t)d * C_DIM + k0 + seg * 8,
                      &Abuf[buf][tile][row][seg * 8]);
    }
    // B: 96 cols x 32 K from xbf[N,P,C] (contiguous over K) -> async b128
    for (int idx = tid; idx < COLS * 4; idx += 384) {
      int col = idx >> 2, seg = idx & 3;
      int ni = col / P_DIM, p = col - ni * P_DIM;
      async_copy_b128(
          xbf + ((size_t)(n0 + ni) * P_DIM + p) * C_DIM + k0 + seg * 8,
          &Bbuf[buf][col][seg * 8]);
    }
  };

  v8f acc[6] = {};

  load_tiles(0, 0);
  wait_async0();
  __syncthreads();

  for (int step = 0; step < C_DIM / 32; ++step) {
    const int cur = step & 1;
    if (step + 1 < C_DIM / 32) load_tiles(step + 1, cur ^ 1);

    // Load A fragment + ALL six B fragments first, then issue six WMMAs
    // back-to-back (no per-WMMA LDS stall; partial dscnt waits possible).
    Frag a, b[6];
    a.u[0] = *reinterpret_cast<const uint4*>(&Abuf[cur][wave][arow][kbA]);
    a.u[1] = *reinterpret_cast<const uint4*>(&Abuf[cur][wave][arow][kbA + 16]);
#pragma unroll
    for (int nt = 0; nt < 6; ++nt) {
      const unsigned short* bp = &Bbuf[cur][nt * 16 + (lane & 15)][kbB];
      b[nt].u[0] = *reinterpret_cast<const uint4*>(bp);
      b[nt].u[1] = *reinterpret_cast<const uint4*>(bp + 16);
    }
#pragma unroll
    for (int nt = 0; nt < 6; ++nt) {
      acc[nt] = __builtin_amdgcn_wmma_f32_16x16x32_bf16(
          false, a.v, false, b[nt].v, (short)0, acc[nt], false, false);
    }
    wait_async0();      // next buffer's async copies complete
    __syncthreads();    // all waves' LDS traffic ordered
  }

  // spill accumulators (+bias) to LDS: D layout VGPR r -> M=r+8*(lane>>4)
  {
    int rbase = mt * 16 + 8 * (lane >> 4);
#pragma unroll
    for (int nt = 0; nt < 6; ++nt) {
      int col = nt * 16 + (lane & 15);
#pragma unroll
      for (int r = 0; r < 8; ++r) {
        int f = rbase + r;
        qkv[mW][f][col] = acc[nt][r] + biasl[mW][f];
      }
    }
  }
  __syncthreads();

  // scores[n,p,p'] = q.k / sqrt(FD)
  for (int idx = tid; idx < NB * P_DIM * P_DIM; idx += 384) {
    int n = idx / 36, rem = idx - n * 36;
    int p = rem / 6, pq = rem - p * 6;
    int ca = n * P_DIM + p, cb = n * P_DIM + pq;
    float s = 0.f;
    for (int f = 0; f < FD_DIM; ++f) s += qkv[0][f][ca] * qkv[1][f][cb];
    sc[n][p][pq] = s * 0.125f;     // 1/sqrt(64)
  }
  __syncthreads();

  // softmax over last dim (96 rows)
  if (tid < NB * P_DIM) {
    int n = tid / P_DIM, p = tid - n * P_DIM;
    float mx = sc[n][p][0];
#pragma unroll
    for (int j = 1; j < 6; ++j) mx = fmaxf(mx, sc[n][p][j]);
    float e[6], sum = 0.f;
#pragma unroll
    for (int j = 0; j < 6; ++j) { e[j] = __expf(sc[n][p][j] - mx); sum += e[j]; }
    float inv = 1.f / sum;
#pragma unroll
    for (int j = 0; j < 6; ++j) sc[n][p][j] = e[j] * inv;
  }
  __syncthreads();

  // AO[col,f] = attn @ V  -> staged bf16 in LDS (conflict-padded rows)
  for (int idx = tid; idx < FD_DIM * COLS; idx += 384) {
    int f = idx / COLS, col = idx - f * COLS;
    int n = col / P_DIM, p = col - n * P_DIM;
    float o = 0.f;
#pragma unroll
    for (int j = 0; j < 6; ++j) o += sc[n][p][j] * qkv[2][f][n * P_DIM + j];
    aot[col][f] = f2bf(o);
  }
  __syncthreads();

  // coalesced b128 stores: ao layout [N,P,D], d = g*64 + f
  for (int idx = tid; idx < COLS * 8; idx += 384) {
    int col = idx >> 3, seg = idx & 7;
    int ni = col / P_DIM, p = col - ni * P_DIM;
    *reinterpret_cast<uint4*>(
        ao + ((size_t)(n0 + ni) * P_DIM + p) * D_DIM + g * FD_DIM + seg * 8) =
        *reinterpret_cast<const uint4*>(&aot[col][seg * 8]);
  }
}

// ---------------- K2: output projection + bias + residual ------------------
// grid = (N/16, C/128), block = 256 (8 waves; wave w -> M-tile w)
__global__ __launch_bounds__(256) void attpcb_out(
    const float* __restrict__ x, const unsigned short* __restrict__ wobf,
    const unsigned short* __restrict__ ao,  // bf16 [N,P,D]
    const float* __restrict__ bo,
    float* __restrict__ out)
{
  __shared__ __align__(16) unsigned short Abuf[2][8][16][32];
  __shared__ __align__(16) unsigned short Bbuf[2][COLS][32];

  const int tid  = threadIdx.x;
  const int wave = tid >> 5;
  const int lane = tid & 31;
  const int n0   = blockIdx.x * NB;
  const int c0   = blockIdx.y * 128;
  const int arow = lane & 15;
  const int kbA  = (lane >> 4) * 8;
  const int kbB  = (lane >> 4) * 8;

  auto load_tiles = [&](int step, int buf) {
    const int k0 = step * 32;
    // A: wo rows, contiguous bf16 -> async b128
    for (int idx = tid; idx < 8 * 16 * 4; idx += 256) {
      int tile = idx >> 6, rem = idx & 63, row = rem >> 2, seg = rem & 3;
      int c = c0 + tile * 16 + row;
      async_copy_b128(wobf + (size_t)c * D_DIM + k0 + seg * 8,
                      &Abuf[buf][tile][row][seg * 8]);
    }
    // B: AO[N,P,D] columns contiguous over K -> async b128
    for (int idx = tid; idx < COLS * 4; idx += 256) {
      int col = idx >> 2, seg = idx & 3;
      int ni = col / P_DIM, p = col - ni * P_DIM;
      async_copy_b128(ao + ((size_t)(n0 + ni) * P_DIM + p) * D_DIM + k0 + seg * 8,
                      &Bbuf[buf][col][seg * 8]);
    }
  };

  v8f acc[6] = {};

  load_tiles(0, 0);
  wait_async0();
  __syncthreads();

  for (int step = 0; step < D_DIM / 32; ++step) {
    const int cur = step & 1;
    if (step + 1 < D_DIM / 32) load_tiles(step + 1, cur ^ 1);

    Frag a, b[6];
    a.u[0] = *reinterpret_cast<const uint4*>(&Abuf[cur][wave][arow][kbA]);
    a.u[1] = *reinterpret_cast<const uint4*>(&Abuf[cur][wave][arow][kbA + 16]);
#pragma unroll
    for (int nt = 0; nt < 6; ++nt) {
      const unsigned short* bp = &Bbuf[cur][nt * 16 + (lane & 15)][kbB];
      b[nt].u[0] = *reinterpret_cast<const uint4*>(bp);
      b[nt].u[1] = *reinterpret_cast<const uint4*>(bp + 16);
    }
#pragma unroll
    for (int nt = 0; nt < 6; ++nt) {
      acc[nt] = __builtin_amdgcn_wmma_f32_16x16x32_bf16(
          false, a.v, false, b[nt].v, (short)0, acc[nt], false, false);
    }
    wait_async0();
    __syncthreads();
  }

  // epilogue: out = x + o + bo
  int rbase = wave * 16 + 8 * (lane >> 4);
#pragma unroll
  for (int nt = 0; nt < 6; ++nt) {
    int col = nt * 16 + (lane & 15);
    int ni = col / P_DIM, p = col - ni * P_DIM;
#pragma unroll
    for (int r = 0; r < 8; ++r) {
      int c = c0 + rbase + r;
      size_t off = ((size_t)(n0 + ni) * C_DIM + c) * P_DIM + p;
      out[off] = x[off] + acc[nt][r] + bo[c];
    }
  }
}

// ---------------------------------------------------------------------------
extern "C" void kernel_launch(void* const* d_in, const int* in_sizes, int n_in,
                              void* d_out, int out_size, void* d_ws, size_t ws_size,
                              hipStream_t stream) {
  const float* x  = (const float*)d_in[0];
  const float* wq = (const float*)d_in[1];
  const float* bq = (const float*)d_in[2];
  const float* wk = (const float*)d_in[3];
  const float* bk = (const float*)d_in[4];
  const float* wv = (const float*)d_in[5];
  const float* bv = (const float*)d_in[6];
  const float* wo = (const float*)d_in[7];
  const float* bo = (const float*)d_in[8];
  float* out = (float*)d_out;

  unsigned short* wbf = (unsigned short*)d_ws;              // 4*DC
  unsigned short* ao  = wbf + (size_t)4 * DC;               // N*P*D
  unsigned short* xbf = ao + (size_t)N_DIM * P_DIM * D_DIM; // N*P*C
  (void)in_sizes; (void)n_in; (void)out_size; (void)ws_size;

  cvt_weights<<<2048, 256, 0, stream>>>(wq, wk, wv, wo, wbf);
  cvt_x<<<N_DIM, 256, 0, stream>>>(x, xbf);

  dim3 g1(N_DIM / NB, G_DIM);
  attpcb_qkv_attn<<<g1, 384, 0, stream>>>(xbf, wbf, bq, bk, bv, ao);

  dim3 g2(N_DIM / NB, C_DIM / 128);
  attpcb_out<<<g2, 256, 0, stream>>>(x, wbf + (size_t)3 * DC, ao, bo, out);
}
